// LstmClassifier_60327110640308
// MI455X (gfx1250) — compile-verified, matrix operand order checked by reading, not run on hardware
//
#include <hip/hip_runtime.h>
#include <hip/hip_bf16.h>
#include <math.h>

#define B_       2048
#define IN_DIM   1024
#define HID      1024
#define T_STEPS  32
#define OUT_DIM  512

typedef __attribute__((ext_vector_type(16))) __bf16 v16bf;
typedef __attribute__((ext_vector_type(8)))  float  v8f;
typedef __attribute__((ext_vector_type(4)))  unsigned int u32x4;

// CDNA5 async global->LDS DMA path (ASYNCcnt) if this toolchain exposes it.
#if defined(__has_builtin)
#  if __has_builtin(__builtin_amdgcn_global_load_async_to_lds_b128)
#    define HAVE_ASYNC_LDS 1
#  endif
#endif
#ifndef HAVE_ASYNC_LDS
#  define HAVE_ASYNC_LDS 0
#endif

#if HAVE_ASYNC_LDS
// builtin signature (from toolchain diagnostic): param0 = int4-vector ptr in
// global addrspace(1), param1 = int4-vector ptr in LDS addrspace(3), then
// imm offset, imm cpol.
typedef int v4i __attribute__((ext_vector_type(4)));
typedef __attribute__((address_space(1))) v4i v4i_g;
typedef __attribute__((address_space(3))) v4i v4i_l;
#endif

__device__ __forceinline__ unsigned short f32_to_bf16_rne(float x) {
    unsigned int u = __float_as_uint(x);
    unsigned int r = u + 0x7FFFu + ((u >> 16) & 1u);
    return (unsigned short)(r >> 16);
}

// ---------------- conversion / prep kernels ----------------
__global__ void k_cvt_bf16(const float* __restrict__ in, unsigned short* __restrict__ out, int n) {
    int i = blockIdx.x * blockDim.x + threadIdx.x;
    if (i < n) out[i] = f32_to_bf16_rne(in[i]);
}

__global__ void k_add_cvt_bf16(const float* __restrict__ a, const float* __restrict__ b,
                               unsigned short* __restrict__ out, int n) {
    int i = blockIdx.x * blockDim.x + threadIdx.x;
    if (i < n) out[i] = f32_to_bf16_rne(a[i] + b[i]);
}

__global__ void k_add_f32(const float* __restrict__ a, const float* __restrict__ b,
                          float* __restrict__ out, int n) {
    int i = blockIdx.x * blockDim.x + threadIdx.x;
    if (i < n) out[i] = a[i] + b[i];
}

// ---------------- WMMA GEMM: C[M,N] = act(A[M,K] @ W[N,K]^T + bias) ----------------
// A, W are bf16 row-major; C is f32 or bf16. M,N from grid (block tile 128x128), K param.
// Double-buffered LDS; async DMA staging overlaps WMMA compute.
#define BM 128
#define BN 128
#define BK 32
#define LDSS 40   // padded row stride in halves (80 bytes, keeps 16B alignment)

template<int RELU, int OUTBF16>
__global__ __launch_bounds__(256)
void k_gemm(const unsigned short* __restrict__ A, int lda,
            const unsigned short* __restrict__ W, int ldw,
            const float* __restrict__ bias,
            void* __restrict__ Cout, int ldc, int K)
{
    __shared__ __align__(16) unsigned short As[2][BM * LDSS];
    __shared__ __align__(16) unsigned short Ws[2][BN * LDSS];

    const int tid  = threadIdx.x;
    const int wave = tid >> 5;
    const int lane = tid & 31;
    const int hf   = lane >> 4;   // lane half selects K sub-chunks / M sub-rows
    const int lrow = lane & 15;
    const int wm   = wave & 3;    // 4 wave-rows  (32 M each)
    const int wn   = wave >> 2;   // 2 wave-cols  (64 N each)

    const int blockM = blockIdx.y * BM;
    const int blockN = blockIdx.x * BN;

    v8f acc[2][4];
#pragma unroll
    for (int i = 0; i < 2; ++i)
#pragma unroll
        for (int j = 0; j < 4; ++j) acc[i][j] = (v8f){};

    const unsigned short* Ablk = A + (size_t)blockM * lda;
    const unsigned short* Wblk = W + (size_t)blockN * ldw;

    // each thread stages two 8-element (16B) chunks per tile: rows r and r+64
    const int sr = tid >> 2;            // 0..63
    const int sc = (tid & 3) << 3;      // 0,8,16,24

    auto stage = [&](int k0, int buf) {
#if HAVE_ASYNC_LDS
#pragma unroll
        for (int it = 0; it < 2; ++it) {
            int rr = sr + it * 64;
            __builtin_amdgcn_global_load_async_to_lds_b128(
                (v4i_g*)(Ablk + (size_t)rr * lda + k0 + sc),
                (v4i_l*)(&As[buf][rr * LDSS + sc]), 0, 0);
            __builtin_amdgcn_global_load_async_to_lds_b128(
                (v4i_g*)(Wblk + (size_t)rr * ldw + k0 + sc),
                (v4i_l*)(&Ws[buf][rr * LDSS + sc]), 0, 0);
        }
#else
#pragma unroll
        for (int it = 0; it < 2; ++it) {
            int rr = sr + it * 64;
            u32x4 va = *(const u32x4*)(Ablk + (size_t)rr * lda + k0 + sc);
            u32x4 vw = *(const u32x4*)(Wblk + (size_t)rr * ldw + k0 + sc);
            *(u32x4*)(&As[buf][rr * LDSS + sc]) = va;
            *(u32x4*)(&Ws[buf][rr * LDSS + sc]) = vw;
        }
        if (k0 + BK < K) {
            __builtin_prefetch(Ablk + (size_t)sr * lda + k0 + BK, 0, 1);
            __builtin_prefetch(Wblk + (size_t)sr * ldw + k0 + BK, 0, 1);
        }
#endif
    };

    auto wait_stage = [&]() {
#if HAVE_ASYNC_LDS
        asm volatile("s_wait_asynccnt 0" ::: "memory");
#endif
    };

    // prologue: stage tile 0
    stage(0, 0);
    wait_stage();
    __syncthreads();

    int cur = 0;
    for (int k0 = 0; k0 < K; k0 += BK) {
        // kick off DMA/loads for the next tile into the other buffer
        if (k0 + BK < K) stage(k0 + BK, cur ^ 1);

        // fragment loads per ISA 16-bit operand layout:
        // lane<16 holds K 0-7 & 16-23 of its row; lane>=16 holds K 8-15 & 24-31.
        const int kc = hf * 8;
        v16bf afrag[2], wfrag[4];
#pragma unroll
        for (int mi = 0; mi < 2; ++mi) {
            const unsigned short* p = &As[cur][(wm * 32 + mi * 16 + lrow) * LDSS];
            u32x4* d = (u32x4*)&afrag[mi];
            d[0] = *(const u32x4*)(p + kc);
            d[1] = *(const u32x4*)(p + kc + 16);
        }
#pragma unroll
        for (int ni = 0; ni < 4; ++ni) {
            const unsigned short* p = &Ws[cur][(wn * 64 + ni * 16 + lrow) * LDSS];
            u32x4* d = (u32x4*)&wfrag[ni];
            d[0] = *(const u32x4*)(p + kc);
            d[1] = *(const u32x4*)(p + kc + 16);
        }

#pragma unroll
        for (int mi = 0; mi < 2; ++mi)
#pragma unroll
            for (int ni = 0; ni < 4; ++ni)
                acc[mi][ni] = __builtin_amdgcn_wmma_f32_16x16x32_bf16(
                    false, afrag[mi], false, wfrag[ni],
                    (short)0, acc[mi][ni], false, false);

        wait_stage();       // next tile fully in LDS (async path)
        __syncthreads();    // + visible to all waves
        cur ^= 1;
    }

    // epilogue: C layout -> lane<16: M=r, N=lane; lane>=16: M=8+r, N=lane-16
#pragma unroll
    for (int mi = 0; mi < 2; ++mi) {
        int rowbase = blockM + wm * 32 + mi * 16 + hf * 8;
#pragma unroll
        for (int ni = 0; ni < 4; ++ni) {
            int n = blockN + wn * 64 + ni * 16 + lrow;
            float bv = bias ? bias[n] : 0.0f;
#pragma unroll
            for (int r = 0; r < 8; ++r) {
                float v = acc[mi][ni][r] + bv;
                if (RELU) v = v > 0.0f ? v : 0.0f;
                size_t off = (size_t)(rowbase + r) * (size_t)ldc + n;
                if (OUTBF16) ((unsigned short*)Cout)[off] = f32_to_bf16_rne(v);
                else         ((float*)Cout)[off] = v;
            }
        }
    }
}

// ---------------- LSTM elementwise cell ----------------
__global__ void k_lstm_cell(const float* __restrict__ gates,
                            float* __restrict__ c,
                            unsigned short* __restrict__ h,
                            int first)
{
    int i = blockIdx.x * blockDim.x + threadIdx.x;
    if (i >= B_ * HID) return;
    int b = i >> 10;          // / HID
    int j = i & (HID - 1);
    const float* g = gates + (size_t)b * (4 * HID);
    float ig = g[j];
    float fg = g[HID + j];
    float gg = g[2 * HID + j];
    float og = g[3 * HID + j];
    float is = 1.0f / (1.0f + __expf(-ig));
    float fs = 1.0f / (1.0f + __expf(-fg));
    float gt = tanhf(gg);
    float os = 1.0f / (1.0f + __expf(-og));
    float cold = first ? 0.0f : c[i];
    float cn = fs * cold + is * gt;
    float hn = os * tanhf(cn);
    c[i] = cn;
    h[i] = f32_to_bf16_rne(hn);
}

// ---------------- orchestration ----------------
extern "C" void kernel_launch(void* const* d_in, const int* in_sizes, int n_in,
                              void* d_out, int out_size, void* d_ws, size_t ws_size,
                              hipStream_t stream) {
    (void)in_sizes; (void)n_in; (void)out_size; (void)ws_size;
    const float* x     = (const float*)d_in[0];
    const float* W_in  = (const float*)d_in[1];
    const float* b_in  = (const float*)d_in[2];
    const float* W_ih  = (const float*)d_in[3];
    const float* b_ih  = (const float*)d_in[4];
    const float* W_hh  = (const float*)d_in[5];
    const float* b_hh  = (const float*)d_in[6];
    const float* W_out = (const float*)d_in[7];
    const float* b_out = (const float*)d_in[8];
    float* out = (float*)d_out;

    char* ws = (char*)d_ws;
    size_t off = 0;
    auto alloc = [&](size_t bytes) -> void* {
        void* p = ws + off;
        off = (off + bytes + 255) & ~(size_t)255;
        return p;
    };
    unsigned short* xb    = (unsigned short*)alloc((size_t)B_ * IN_DIM * 2);
    unsigned short* Winb  = (unsigned short*)alloc((size_t)HID * IN_DIM * 2);
    unsigned short* Wcb   = (unsigned short*)alloc((size_t)4 * HID * HID * 2);
    unsigned short* Woutb = (unsigned short*)alloc((size_t)OUT_DIM * HID * 2);
    float*          bcomb = (float*)alloc((size_t)4 * HID * 4);
    unsigned short* hb    = (unsigned short*)alloc((size_t)B_ * HID * 2);
    float*          cbuf  = (float*)alloc((size_t)B_ * HID * 4);
    float*          gates = (float*)alloc((size_t)B_ * 4 * HID * 4);

    const int THR = 256;
    auto nb = [](long long n) { return (unsigned)((n + 255) / 256); };

    // one-time (per call) precision conversion of inputs/weights
    k_cvt_bf16<<<nb((long long)B_ * IN_DIM), THR, 0, stream>>>(x, xb, B_ * IN_DIM);
    k_cvt_bf16<<<nb((long long)HID * IN_DIM), THR, 0, stream>>>(W_in, Winb, HID * IN_DIM);
    k_cvt_bf16<<<nb((long long)OUT_DIM * HID), THR, 0, stream>>>(W_out, Woutb, OUT_DIM * HID);
    k_add_cvt_bf16<<<nb((long long)4 * HID * HID), THR, 0, stream>>>(W_ih, W_hh, Wcb, 4 * HID * HID);
    k_add_f32<<<nb(4 * HID), THR, 0, stream>>>(b_ih, b_hh, bcomb, 4 * HID);

    // h0 = relu(x @ W_in^T + b_in)  -> bf16
    k_gemm<1, 1><<<dim3(HID / BN, B_ / BM), THR, 0, stream>>>(
        xb, IN_DIM, Winb, IN_DIM, b_in, hb, HID, IN_DIM);

    for (int t = 0; t < T_STEPS; ++t) {
        // gates = h @ (W_ih + W_hh)^T + (b_ih + b_hh)  -> f32
        k_gemm<0, 0><<<dim3(4 * HID / BN, B_ / BM), THR, 0, stream>>>(
            hb, HID, Wcb, HID, bcomb, gates, 4 * HID, HID);
        // elementwise cell: updates c, writes h_t (bf16)
        k_lstm_cell<<<nb((long long)B_ * HID), THR, 0, stream>>>(gates, cbuf, hb, t == 0);
        // out[:, t, :] = h_t @ W_out^T + b_out  (fused per-step output projection)
        k_gemm<0, 0><<<dim3(OUT_DIM / BN, B_ / BM), THR, 0, stream>>>(
            hb, HID, Woutb, HID, b_out,
            out + (size_t)t * OUT_DIM, T_STEPS * OUT_DIM, HID);
    }
}
